// MultiHead_Attention_73065983640334
// MI455X (gfx1250) — matrix operand order repaired and checked by reading, not compiled, not run on hardware
//
#include <hip/hip_runtime.h>

typedef __attribute__((ext_vector_type(16))) __bf16       v16bf;
typedef __attribute__((ext_vector_type(8)))  float        v8f;
typedef __attribute__((ext_vector_type(4)))  unsigned int u32x4;

union Frag { u32x4 q[2]; v16bf v; };
static_assert(sizeof(Frag) == 32, "frag size");

#define DEV __device__ __forceinline__

DEV v8f wmma_bf16(v16bf a, v16bf b, v8f c) {
  // D = A(16x32 bf16) * B(32x16 bf16) + C(16x16 f32)
  return __builtin_amdgcn_wmma_f32_16x16x32_bf16(false, a, false, b, (short)0, c,
                                                 false, false);
}

DEV v8f v8f_zero() {
  v8f z = {0.f, 0.f, 0.f, 0.f, 0.f, 0.f, 0.f, 0.f};
  return z;
}

// ---- problem sizes (fixed) ----
// B=2, S=2048, D=1024, H=16, DH=64; rows = B*S = 4096

// ---- workspace layout, offsets in bf16 elements (total 24M elems = 48 MB) ----
static constexpr size_t OFF_XBF = 0;                  // [4096][1024] bf16 copy of x
static constexpr size_t OFF_WQT = (size_t)4 << 20;    // [n][k] bf16, pre-scaled 0.125
static constexpr size_t OFF_WKT = (size_t)5 << 20;    // [n][k] bf16
static constexpr size_t OFF_WVT = (size_t)6 << 20;    // [n][k] bf16
static constexpr size_t OFF_WOT = (size_t)7 << 20;    // [n][k] bf16
static constexpr size_t OFF_Q   = (size_t)8 << 20;    // [bh][s][dh]  (Q pre-scaled)
static constexpr size_t OFF_K   = (size_t)12 << 20;   // [bh][s][dh]
static constexpr size_t OFF_VT  = (size_t)16 << 20;   // [bh][dh][s]  (transposed)
static constexpr size_t OFF_CTX = (size_t)20 << 20;   // [4096][1024]

// ---------------- kernel 1: convert + transpose ----------------
__global__ __launch_bounds__(256) void mha_prep(
    const float* __restrict__ x, const float* __restrict__ Wq,
    const float* __restrict__ Wk, const float* __restrict__ Wv,
    const float* __restrict__ Wo, __bf16* __restrict__ ws) {
  unsigned i = blockIdx.x * 256u + threadIdx.x;  // [0, 4M)
  ws[OFF_XBF + i] = (__bf16)x[i];
  if (i < (1u << 20)) {
    unsigned n = i >> 10, k = i & 1023u;
    unsigned src = k * 1024u + n;                // transpose: Wt[n][k] = W[k][n]
    ws[OFF_WQT + i] = (__bf16)(Wq[src] * 0.125f);  // fold 1/sqrt(DH)
    ws[OFF_WKT + i] = (__bf16)(Wk[src]);
    ws[OFF_WVT + i] = (__bf16)(Wv[src]);
    ws[OFF_WOT + i] = (__bf16)(Wo[src]);
  }
}

// ---- fragment loaders: base pointer + compile-time offset -> imm-offset loads ----
// A-frag 16x32 bf16: lane M=l15, elems e -> K = k + 16*hh + e (16 contiguous)
// two row-blocks, 16 rows (16*1024 elems) apart
DEV void load_a2(Frag a[2], const __bf16* pa, int k) {
  a[0].q[0] = *(const u32x4*)(pa + k);
  a[0].q[1] = *(const u32x4*)(pa + k + 8);
  a[1].q[0] = *(const u32x4*)(pa + 16 * 1024 + k);
  a[1].q[1] = *(const u32x4*)(pa + 16 * 1024 + k + 8);
}
// B-frag 32x16 bf16: lane N=l15, elems e -> K = k + 16*hh + e (16 contiguous)
// four col-tiles, 16 rows (16*1024 elems) apart
DEV void load_b4(Frag b[4], const __bf16* pb, int k) {
#pragma unroll
  for (int t = 0; t < 4; ++t) {
    b[t].q[0] = *(const u32x4*)(pb + t * 16 * 1024 + k);
    b[t].q[1] = *(const u32x4*)(pb + t * 16 * 1024 + k + 8);
  }
}

// ---------------- shared GEMM core: one wave -> 32x64 tile, K=1024 ----------------
// fully unrolled, explicitly double-buffered (loads one stage ahead of WMMAs)
DEV void gemm_32x64(const __bf16* __restrict__ A, const __bf16* __restrict__ Bt,
                    int r0, int n0, int l15, int hh, v8f acc[2][4]) {
  const __bf16* pa = A + (size_t)(r0 + l15) * 1024 + 16 * hh;
  const __bf16* pb = Bt + (size_t)(n0 + l15) * 1024 + 16 * hh;

  Frag a0[2], b0[4], a1[2], b1[4];
  load_a2(a0, pa, 0);
  load_b4(b0, pb, 0);
#pragma unroll
  for (int kc = 0; kc < 1024; kc += 64) {
    load_a2(a1, pa, kc + 32);   // stage-1 fragments in flight ...
    load_b4(b1, pb, kc + 32);
#pragma unroll
    for (int t = 0; t < 4; ++t) {  // ... while stage-0 computes
      acc[0][t] = wmma_bf16(a0[0].v, b0[t].v, acc[0][t]);
      acc[1][t] = wmma_bf16(a0[1].v, b0[t].v, acc[1][t]);
    }
    if (kc + 64 < 1024) {
      load_a2(a0, pa, kc + 64);
      load_b4(b0, pb, kc + 64);
    }
#pragma unroll
    for (int t = 0; t < 4; ++t) {
      acc[0][t] = wmma_bf16(a1[0].v, b1[t].v, acc[0][t]);
      acc[1][t] = wmma_bf16(a1[1].v, b1[t].v, acc[1][t]);
    }
  }
}

// ---------------- kernel 2: QKV projections ----------------
__global__ __launch_bounds__(256) void mha_qkv(__bf16* __restrict__ ws) {
  const int w = threadIdx.x >> 5, lane = threadIdx.x & 31;
  const int l15 = lane & 15, hh = lane >> 4;
  const int proj = blockIdx.z;  // 0=Q, 1=K, 2=V
  const __bf16* X = ws + OFF_XBF;
  const __bf16* Wt = ws + (proj == 0 ? OFF_WQT : (proj == 1 ? OFF_WKT : OFF_WVT));
  __bf16* dst = ws + (proj == 0 ? OFF_Q : (proj == 1 ? OFF_K : OFF_VT));
  const int r0 = blockIdx.x * 128 + (w & 3) * 32;
  const int n0 = blockIdx.y * 128 + (w >> 2) * 64;

  v8f acc[2][4];
#pragma unroll
  for (int rb = 0; rb < 2; ++rb)
#pragma unroll
    for (int t = 0; t < 4; ++t) acc[rb][t] = v8f_zero();

  gemm_32x64(X, Wt, r0, n0, l15, hh, acc);

  // C-layout: elem r -> row M = r + 8*hh, col N = l15
#pragma unroll
  for (int rb = 0; rb < 2; ++rb)
#pragma unroll
    for (int t = 0; t < 4; ++t) {
      const int n = n0 + 16 * t + l15;
      const int h = n >> 6, dh = n & 63;
#pragma unroll
      for (int r = 0; r < 8; ++r) {
        const int row = r0 + 16 * rb + r + 8 * hh;
        const int b = row >> 11, s = row & 2047;
        const int bh = b * 16 + h;
        const size_t idx = (proj == 2)
                               ? (((size_t)bh * 64 + dh) * 2048 + s)   // V transposed
                               : (((size_t)bh * 2048 + s) * 64 + dh);  // Q, K
        dst[idx] = (__bf16)acc[rb][t][r];
      }
    }
}

// ---------------- kernel 3: causal flash attention ----------------
__global__ __launch_bounds__(256) void mha_attn(__bf16* __restrict__ ws) {
  __shared__ __align__(16) __bf16 Pbuf[8][16][32];  // per-wave P transpose buffer

  const int w = threadIdx.x >> 5, lane = threadIdx.x & 31;
  const int l15 = lane & 15, hh = lane >> 4;
  const int bh = blockIdx.x & 31;            // b*16 + h
  const int q0 = (blockIdx.x >> 5) * 128 + w * 16;

  const __bf16* Qb = ws + OFF_Q + (size_t)bh * 2048 * 64;
  // per-lane base pointers; loop advances them, loads use immediate offsets
  const __bf16* pk = ws + OFF_K + (size_t)bh * 2048 * 64 + (size_t)l15 * 64 + 16 * hh;
  const __bf16* pv = ws + OFF_VT + (size_t)bh * 64 * 2048 + (size_t)l15 * 2048 + 16 * hh;

  // resident Q A-frags (dh chunks 0..31 and 32..63), already scaled by 1/8
  Frag aq[2];
#pragma unroll
  for (int c = 0; c < 2; ++c) {
    const __bf16* p = Qb + (size_t)(q0 + l15) * 64 + 32 * c + 16 * hh;
    aq[c].q[0] = *(const u32x4*)(p);
    aq[c].q[1] = *(const u32x4*)(p + 8);
  }

  v8f acc[4];
#pragma unroll
  for (int t = 0; t < 4; ++t) acc[t] = v8f_zero();
  float m_i[8], l_i[8];
#pragma unroll
  for (int r = 0; r < 8; ++r) { m_i[r] = -1e30f; l_i[r] = 0.f; }

  // K fragments double-buffered: prologue load for kb = 0
  // K-frag addr: (kb + 16*jt + l15)*64 + 32*c + 16*hh = pk + kb*64 + jt*1024 + c*32
  Frag bk[2][2];
#pragma unroll
  for (int jt = 0; jt < 2; ++jt)
#pragma unroll
    for (int c = 0; c < 2; ++c) {
      bk[jt][c].q[0] = *(const u32x4*)(pk + jt * 1024 + c * 32);
      bk[jt][c].q[1] = *(const u32x4*)(pk + jt * 1024 + c * 32 + 8);
    }

  for (int kb = 0; kb < q0 + 16; kb += 32) {
    // ---- issue V-frag loads first: hidden behind score WMMAs + softmax ----
    // V-frag addr: (16*t + l15)*2048 + kb + 16*hh = pv + t*32768 (+ kb via bump)
    Frag bv[4];
#pragma unroll
    for (int t = 0; t < 4; ++t) {
      bv[t].q[0] = *(const u32x4*)(pv + t * 32768);
      bv[t].q[1] = *(const u32x4*)(pv + t * 32768 + 8);
    }
    // ---- scores: two 16x16 tiles, contraction over dh=64 (2 WMMAs each) ----
    v8f s[2];
#pragma unroll
    for (int jt = 0; jt < 2; ++jt) {
      v8f c0 = v8f_zero();
      c0 = wmma_bf16(aq[0].v, bk[jt][0].v, c0);
      c0 = wmma_bf16(aq[1].v, bk[jt][1].v, c0);
      s[jt] = c0;
    }
    // ---- causal mask + online softmax (rows spread over 16-lane halves) ----
    const int key0 = kb + l15, key1 = kb + 16 + l15;
    float p0[8], p1[8];
#pragma unroll
    for (int r = 0; r < 8; ++r) {
      const int q = q0 + r + 8 * hh;
      float s0 = (key0 <= q) ? s[0][r] : -1e30f;
      float s1 = (key1 <= q) ? s[1][r] : -1e30f;
      float mx = fmaxf(s0, s1);
      mx = fmaxf(mx, __shfl_xor(mx, 1, 32));
      mx = fmaxf(mx, __shfl_xor(mx, 2, 32));
      mx = fmaxf(mx, __shfl_xor(mx, 4, 32));
      mx = fmaxf(mx, __shfl_xor(mx, 8, 32));
      const float mnew = fmaxf(m_i[r], mx);
      const float alpha = __expf(m_i[r] - mnew);
      const float e0 = __expf(s0 - mnew);
      const float e1 = __expf(s1 - mnew);
      float rs = e0 + e1;
      rs += __shfl_xor(rs, 1, 32);
      rs += __shfl_xor(rs, 2, 32);
      rs += __shfl_xor(rs, 4, 32);
      rs += __shfl_xor(rs, 8, 32);
      l_i[r] = l_i[r] * alpha + rs;
      m_i[r] = mnew;
#pragma unroll
      for (int t = 0; t < 4; ++t) acc[t][r] *= alpha;
      p0[r] = e0;
      p1[r] = e1;
    }
    // ---- double-buffer: issue next step's K-frags during softmax tail ----
    pk += 32 * 64;
    if (kb + 32 < q0 + 16) {  // wave-uniform
#pragma unroll
      for (int jt = 0; jt < 2; ++jt)
#pragma unroll
        for (int c = 0; c < 2; ++c) {
          bk[jt][c].q[0] = *(const u32x4*)(pk + jt * 1024 + c * 32);
          bk[jt][c].q[1] = *(const u32x4*)(pk + jt * 1024 + c * 32 + 8);
        }
    }
    // ---- transpose P (C-layout) -> A-layout through LDS ----
#pragma unroll
    for (int r = 0; r < 8; ++r) {
      const int row = r + 8 * hh;
      Pbuf[w][row][l15] = (__bf16)p0[r];
      Pbuf[w][row][16 + l15] = (__bf16)p1[r];
    }
    asm volatile("s_wait_dscnt 0" ::: "memory");  // intra-wave LDS RAW
    Frag pa;
    pa.q[0] = *(const u32x4*)&Pbuf[w][l15][16 * hh];
    pa.q[1] = *(const u32x4*)&Pbuf[w][l15][16 * hh + 8];
    // ---- acc += P(16x32) @ V(32x64): 4 WMMAs (bv long in flight by now) ----
#pragma unroll
    for (int t = 0; t < 4; ++t) acc[t] = wmma_bf16(pa.v, bv[t].v, acc[t]);
    pv += 32;
    asm volatile("" ::: "memory");  // keep next-iter LDS stores after these loads
  }

  // ---- normalize and store ctx (bf16, [row][d]) ----
  float linv[8];
#pragma unroll
  for (int r = 0; r < 8; ++r) linv[r] = 1.0f / l_i[r];
  const int b = bh >> 4, h = bh & 15;
  __bf16* ctx = ws + OFF_CTX;
#pragma unroll
  for (int t = 0; t < 4; ++t) {
    const int col = h * 64 + 16 * t + l15;
#pragma unroll
    for (int r = 0; r < 8; ++r) {
      const int row = b * 2048 + q0 + r + 8 * hh;
      ctx[(size_t)row * 1024 + col] = (__bf16)(acc[t][r] * linv[r]);
    }
  }
}

// ---------------- kernel 4: output projection + bias ----------------
__global__ __launch_bounds__(256) void mha_oproj(const __bf16* __restrict__ ws,
                                                 const float* __restrict__ bo,
                                                 float* __restrict__ out) {
  const int w = threadIdx.x >> 5, lane = threadIdx.x & 31;
  const int l15 = lane & 15, hh = lane >> 4;
  const int r0 = blockIdx.x * 128 + (w & 3) * 32;
  const int n0 = blockIdx.y * 128 + (w >> 2) * 64;

  v8f acc[2][4];
#pragma unroll
  for (int rb = 0; rb < 2; ++rb)
#pragma unroll
    for (int t = 0; t < 4; ++t) acc[rb][t] = v8f_zero();

  gemm_32x64(ws + OFF_CTX, ws + OFF_WOT, r0, n0, l15, hh, acc);

#pragma unroll
  for (int t = 0; t < 4; ++t) {
    const int n = n0 + 16 * t + l15;
    const float bias = bo[n];
#pragma unroll
    for (int rb = 0; rb < 2; ++rb)
#pragma unroll
      for (int r = 0; r < 8; ++r) {
        const int row = r0 + 16 * rb + r + 8 * hh;
        out[(size_t)row * 1024 + n] = acc[rb][t][r] + bias;
      }
  }
}

// ---------------- host launcher ----------------
extern "C" void kernel_launch(void* const* d_in, const int* in_sizes, int n_in,
                              void* d_out, int out_size, void* d_ws, size_t ws_size,
                              hipStream_t stream) {
  (void)in_sizes; (void)n_in; (void)out_size; (void)ws_size;
  const float* x = (const float*)d_in[0];
  const float* Wq = (const float*)d_in[1];
  const float* Wk = (const float*)d_in[2];
  const float* Wv = (const float*)d_in[3];
  const float* Wo = (const float*)d_in[4];
  const float* bo = (const float*)d_in[5];
  float* out = (float*)d_out;
  __bf16* ws = (__bf16*)d_ws;

  // 1) bf16 conversions + weight transposes (4M threads)
  mha_prep<<<16384, 256, 0, stream>>>(x, Wq, Wk, Wv, Wo, ws);
  // 2) Q/K/V projections: rows 4096 / 128, cols 1024 / 128, z = projection
  mha_qkv<<<dim3(32, 8, 3), 256, 0, stream>>>(ws);
  // 3) flash attention: 32 (b,h) x 16 q-chunks of 128 rows (8 waves x 16 rows)
  mha_attn<<<512, 256, 0, stream>>>(ws);
  // 4) output projection + bias
  mha_oproj<<<dim3(32, 8, 1), 256, 0, stream>>>(ws, bo, out);
}
